// SelfBlock_90452011254118
// MI455X (gfx1250) — matrix operand order repaired and checked by reading, not compiled, not run on hardware
//
#include <hip/hip_runtime.h>
#include <stdint.h>

// Problem constants (match reference)
#define BB 2
#define NN 2048
#define DD 1024
#define HH 16
#define HD 64

typedef __attribute__((ext_vector_type(16))) __bf16 v16bf;
typedef __attribute__((ext_vector_type(8)))  float  v8f;
typedef __attribute__((ext_vector_type(8)))  unsigned int v8u;
typedef int v4i __attribute__((vector_size(16)));   // matches builtin param type

#define ASG __attribute__((address_space(1)))
#define ASL __attribute__((address_space(3)))

#if __has_builtin(__builtin_amdgcn_global_load_async_to_lds_b128)
#define HAVE_ASYNC 1
// async 16B copy global->LDS, tracked by ASYNCcnt
#define COPY16(gp, lp)                                            \
    __builtin_amdgcn_global_load_async_to_lds_b128(               \
        (ASG v4i*)(void*)(const_cast<unsigned short*>(gp)),       \
        (ASL v4i*)(void*)(lp), 0, 0)
#define WAIT_ASYNC(n) asm volatile("s_wait_asynccnt " #n ::: "memory")
#else
#define COPY16(gp, lp) do { *(uint4*)(lp) = *(const uint4*)(gp); } while (0)
#define WAIT_ASYNC(n)
#endif

static __device__ __forceinline__ v8f vzero8() {
    v8f v;
#pragma unroll
    for (int i = 0; i < 8; ++i) v[i] = 0.0f;
    return v;
}

// fp32 -> bf16 bits, round-to-nearest-even
static __device__ __forceinline__ unsigned short f2bf(float f) {
    unsigned int u = __builtin_bit_cast(unsigned int, f);
    unsigned int r = u + 0x7FFFu + ((u >> 16) & 1u);
    return (unsigned short)(r >> 16);
}

static __device__ __forceinline__ v16bf make_frag(uint4 a, uint4 b) {
    v8u u;
    u[0] = a.x; u[1] = a.y; u[2] = a.z; u[3] = a.w;
    u[4] = b.x; u[5] = b.y; u[6] = b.z; u[7] = b.w;
    return __builtin_bit_cast(v16bf, u);
}

// A-matrix fragment (16x32, MxK): lane(0..15)=row M, elems 0..7 = K[8*hi+e],
// elems 8..15 = K[16+8*hi+e]  (ISA 7.12.2 "16-bit A-Matrix 16x32")
static __device__ __forceinline__ v16bf load_afrag(const unsigned short* base,
                                                   int stride, int rowbase, int kbase) {
    int lane = threadIdx.x & 31, lo = lane & 15, hi = lane >> 4;
    const unsigned short* p = base + (size_t)(rowbase + lo) * stride + kbase + 8 * hi;
    uint4 c0 = *(const uint4*)p;
    uint4 c1 = *(const uint4*)(p + 16);
    return make_frag(c0, c1);
}

// B-matrix fragment (32x16, KxN): lane(0..15)=col N, elems 0..7 = K[8*hi+e],
// elems 8..15 = K[16+8*hi+e] (same k order as A); row of the row-major
// source holds column N, contiguous along K.
static __device__ __forceinline__ v16bf load_bfrag(const unsigned short* base,
                                                   int stride, int rowbase, int kbase) {
    int lane = threadIdx.x & 31, lo = lane & 15, hi = lane >> 4;
    const unsigned short* p = base + (size_t)(rowbase + lo) * stride + kbase + 8 * hi;
    uint4 c0 = *(const uint4*)p;
    uint4 c1 = *(const uint4*)(p + 16);
    return make_frag(c0, c1);
}

static __device__ __forceinline__ v8f wmma_bf16(v16bf a, v16bf b, v8f c) {
    return __builtin_amdgcn_wmma_f32_16x16x32_bf16(false, a, false, b, (short)0, c,
                                                   false, false);
}

// per-wave LDS ordering: stores by this wave complete before following loads
static __device__ __forceinline__ void lds_fence_wave() {
    asm volatile("s_wait_dscnt 0" ::: "memory");
}

#define LDSP 40  // padded LDS row stride (elements); 80B keeps b128 conflict-free

// =====================================================================
// Generic bf16 GEMM: C[M,N] = A[M,K] @ W[N,K]^T + bias (+ residual)
// A,W pre-packed bf16. Tile 128x64, BK=32, 128 threads (4 waves x 32 rows).
// Double-buffered LDS filled with async global->LDS b128 copies; the
// K-loop is peeled so steady-state iterations have no conditionals.
// Output fp32 (Cf) or bf16 (Cb).
// =====================================================================
__global__ __launch_bounds__(128) void gemm_kernel(
    const unsigned short* __restrict__ A, int lda,
    const unsigned short* __restrict__ W, int ldw,
    const float* __restrict__ bias,
    const float* __restrict__ resid, int ldr,
    float* __restrict__ Cf, unsigned short* __restrict__ Cb, int ldc,
    int N, int K)
{
    __shared__ __align__(16) unsigned short As[2][128 * LDSP];
    __shared__ __align__(16) unsigned short Ws[2][64 * LDSP];

    const int tiles_n = N >> 6;
    const int tm = (blockIdx.x / tiles_n) << 7;   // 128-row tiles
    const int tn = (blockIdx.x % tiles_n) << 6;   // 64-col tiles
    const int tid = threadIdx.x;
    const int wave = tid >> 5;

    v8f acc[2][4];
#pragma unroll
    for (int a = 0; a < 2; ++a)
#pragma unroll
        for (int f = 0; f < 4; ++f) acc[a][f] = vzero8();

    // stage one 128x32 A tile (512 chunks) + 64x32 W tile (256 chunks);
    // 128 threads -> 4 + 2 b128 copies per thread per K-step.
    auto stage = [&](int buf, int kt) {
#pragma unroll
        for (int j = 0; j < 4; ++j) {
            int cI = tid + j * 128, row = cI >> 2, ch = cI & 3;
            const unsigned short* gp = A + (size_t)(tm + row) * lda + kt + ch * 8;
            unsigned short* lp = &As[buf][row * LDSP + ch * 8];
            COPY16(gp, lp);
        }
#pragma unroll
        for (int j = 0; j < 2; ++j) {
            int cI = tid + j * 128, row = cI >> 2, ch = cI & 3;
            const unsigned short* gp = W + (size_t)(tn + row) * ldw + kt + ch * 8;
            unsigned short* lp = &Ws[buf][row * LDSP + ch * 8];
            COPY16(gp, lp);
        }
    };

    auto compute = [&](int buf) {
        v16bf af0 = load_afrag(As[buf], LDSP, wave * 32, 0);
        v16bf af1 = load_afrag(As[buf], LDSP, wave * 32 + 16, 0);
#pragma unroll
        for (int f = 0; f < 4; ++f) {
            v16bf bf = load_bfrag(Ws[buf], LDSP, f * 16, 0);
            acc[0][f] = wmma_bf16(af0, bf, acc[0][f]);
            acc[1][f] = wmma_bf16(af1, bf, acc[1][f]);
        }
    };

    int cur = 0;
    stage(0, 0);
    for (int kt = 0; kt + 32 < K; kt += 32) {
        stage(cur ^ 1, kt + 32);   // prefetch next tile into other buffer
        WAIT_ASYNC(6);             // >6 outstanding -> tile `cur` still landing
        __syncthreads();
        compute(cur);
        __syncthreads();           // readers done before buffer is overwritten
        cur ^= 1;
    }
    WAIT_ASYNC(0);                 // last tile fully landed
    __syncthreads();
    compute(cur);

    const int lane = tid & 31, lo = lane & 15, hi = lane >> 4;
#pragma unroll
    for (int a = 0; a < 2; ++a) {
#pragma unroll
        for (int f = 0; f < 4; ++f) {
#pragma unroll
            for (int r = 0; r < 8; ++r) {
                int gm = tm + wave * 32 + a * 16 + r + 8 * hi;
                int gn = tn + f * 16 + lo;
                float v = acc[a][f][r] + bias[gn];
                if (resid) v += resid[(size_t)gm * ldr + gn];
                if (Cb) Cb[(size_t)gm * ldc + gn] = f2bf(v);
                else    Cf[(size_t)gm * ldc + gn] = v;
            }
        }
    }
}

// =====================================================================
// fp32 -> bf16 pack with optional row restride (for concat left half).
// One float4 -> uint2 per thread.
// =====================================================================
__global__ __launch_bounds__(256) void pack_bf16_kernel(
    const float* __restrict__ src, unsigned short* __restrict__ dst,
    int ncols, int ldd)
{
    size_t idx4 = (size_t)blockIdx.x * 256 + threadIdx.x;
    int nc4 = ncols >> 2;
    size_t row = idx4 / nc4;
    int c = (int)(idx4 % nc4) << 2;
    float4 v = ((const float4*)src)[idx4];
    uint2 u;
    u.x = (unsigned int)f2bf(v.x) | ((unsigned int)f2bf(v.y) << 16);
    u.y = (unsigned int)f2bf(v.z) | ((unsigned int)f2bf(v.w) << 16);
    *(uint2*)(dst + row * ldd + c) = u;
}

// =====================================================================
// RoPE + repack: qkv fp32 [B,N,(H,HD,3)] -> qb,kb bf16 [B,H,N,HD],
// vT bf16 [B,H,HD,N]. One thread per (b,h,n,pair).
// =====================================================================
__global__ __launch_bounds__(256) void rope_pack_kernel(
    const float* __restrict__ qkv, const float* __restrict__ enc,
    unsigned short* __restrict__ qb, unsigned short* __restrict__ kb,
    unsigned short* __restrict__ vT)
{
    size_t idx = (size_t)blockIdx.x * 256 + threadIdx.x;
    int p = (int)(idx & 31);          // HD/2 = 32 pairs
    size_t t = idx >> 5;
    int n = (int)(t & (NN - 1));
    size_t t2 = t >> 11;              // NN = 2^11
    int h = (int)(t2 & (HH - 1));
    int b = (int)(t2 >> 4);           // HH = 16

    int d0 = 2 * p, d1 = d0 + 1;
    size_t row = (size_t)(b * NN + n) * (3 * DD);
    int c0 = (h * HD + d0) * 3;
    int c1 = c0 + 3;
    float q0 = qkv[row + c0],     q1 = qkv[row + c1];
    float k0 = qkv[row + c0 + 1], k1 = qkv[row + c1 + 1];
    float v0 = qkv[row + c0 + 2], v1 = qkv[row + c1 + 2];

    size_t eoff = (((size_t)b * HH + h) * NN + n) * HD;
    const size_t E2 = (size_t)BB * HH * NN * HD;
    float e00 = enc[eoff + d0],      e01 = enc[eoff + d1];
    float e10 = enc[E2 + eoff + d0], e11 = enc[E2 + eoff + d1];

    float qr0 = q0 * e00 - q1 * e10;
    float qr1 = q1 * e01 + q0 * e11;
    float kr0 = k0 * e00 - k1 * e10;
    float kr1 = k1 * e01 + k0 * e11;

    size_t bh = (size_t)b * HH + h;
    size_t offq = (bh * NN + n) * HD + d0;
    *(unsigned int*)(qb + offq) =
        (unsigned int)f2bf(qr0) | ((unsigned int)f2bf(qr1) << 16);
    *(unsigned int*)(kb + offq) =
        (unsigned int)f2bf(kr0) | ((unsigned int)f2bf(kr1) << 16);

    size_t offv = (bh * HD + d0) * NN + n;
    vT[offv]      = f2bf(v0);
    vT[offv + NN] = f2bf(v1);
}

// =====================================================================
// Flash attention: block = (bh, qtile of 64); 4 waves x 16 q-rows each.
// 32-key tiles: 4 WMMA for S = Q K^T, online softmax, 4 WMMA for O += P V.
// Emits ctx directly as bf16 [B,N,H*HD] for the out-projection GEMM.
// =====================================================================
__global__ __launch_bounds__(128) void attn_kernel(
    const unsigned short* __restrict__ qb, const unsigned short* __restrict__ kb,
    const unsigned short* __restrict__ vT, unsigned short* __restrict__ ctxb)
{
    const int bh   = blockIdx.x;
    const int q0   = blockIdx.y * 64;
    const int wave = threadIdx.x >> 5;
    const int lane = threadIdx.x & 31, lo = lane & 15, hi = lane >> 4;

    const unsigned short* qbase = qb + (size_t)bh * NN * HD;
    const unsigned short* kbase = kb + (size_t)bh * NN * HD;
    const unsigned short* vbase = vT + (size_t)bh * HD * NN;

    __shared__ __align__(16) unsigned short Ps[4][16 * LDSP];

    v16bf qf0 = load_afrag(qbase, HD, q0 + wave * 16, 0);
    v16bf qf1 = load_afrag(qbase, HD, q0 + wave * 16, 32);

    float Mrow[8], Lrow[8];
    v8f O[4];
#pragma unroll
    for (int r = 0; r < 8; ++r) { Mrow[r] = -1e30f; Lrow[r] = 0.0f; }
#pragma unroll
    for (int f = 0; f < 4; ++f) O[f] = vzero8();

    const float scale = 0.125f;  // 1/sqrt(64)

    for (int kt = 0; kt < NN; kt += 32) {
        v8f S0 = vzero8(), S1 = vzero8();
        v16bf k0a = load_bfrag(kbase, HD, kt,      0);
        v16bf k0b = load_bfrag(kbase, HD, kt,      32);
        v16bf k1a = load_bfrag(kbase, HD, kt + 16, 0);
        v16bf k1b = load_bfrag(kbase, HD, kt + 16, 32);
        S0 = wmma_bf16(qf0, k0a, S0);
        S0 = wmma_bf16(qf1, k0b, S0);
        S1 = wmma_bf16(qf0, k1a, S1);
        S1 = wmma_bf16(qf1, k1b, S1);

        float corr[8];
#pragma unroll
        for (int r = 0; r < 8; ++r) {
            float s0 = S0[r] * scale, s1 = S1[r] * scale;
            float mx = fmaxf(s0, s1);
#pragma unroll
            for (int m = 1; m <= 8; m <<= 1) mx = fmaxf(mx, __shfl_xor(mx, m, 32));
            float Mnew = fmaxf(Mrow[r], mx);
            float p0 = __expf(s0 - Mnew);
            float p1 = __expf(s1 - Mnew);
            float ps = p0 + p1;
#pragma unroll
            for (int m = 1; m <= 8; m <<= 1) ps += __shfl_xor(ps, m, 32);
            float cr = __expf(Mrow[r] - Mnew);
            Lrow[r] = Lrow[r] * cr + ps;
            Mrow[r] = Mnew;
            corr[r] = cr;
            // P: score layout -> A layout via per-wave LDS scratch
            Ps[wave][(r + 8 * hi) * LDSP + lo]      = f2bf(p0);
            Ps[wave][(r + 8 * hi) * LDSP + 16 + lo] = f2bf(p1);
        }
        lds_fence_wave();
        v16bf pf = load_afrag(Ps[wave], LDSP, 0, 0);
#pragma unroll
        for (int f = 0; f < 4; ++f) {
#pragma unroll
            for (int r = 0; r < 8; ++r) O[f][r] *= corr[r];
            v16bf vf = load_bfrag(vbase, NN, f * 16, kt);  // vT rows = dims
            O[f] = wmma_bf16(pf, vf, O[f]);
        }
        lds_fence_wave();
    }

    const int b = bh >> 4, h = bh & 15;
#pragma unroll
    for (int f = 0; f < 4; ++f) {
#pragma unroll
        for (int r = 0; r < 8; ++r) {
            int gq = q0 + wave * 16 + r + 8 * hi;
            int d  = f * 16 + lo;
            float v = O[f][r] * (1.0f / Lrow[r]);
            ctxb[((size_t)(b * NN + gq)) * DD + h * HD + d] = f2bf(v);
        }
    }
}

// =====================================================================
// Row LayerNorm + exact GELU over [B*N, 2D] fp32 -> bf16. Block per row.
// Thread owns 8 consecutive columns -> one b128 bf16 store.
// =====================================================================
__global__ __launch_bounds__(256) void ln_gelu_kernel(
    const float* __restrict__ h, unsigned short* __restrict__ hb,
    const float* __restrict__ g, const float* __restrict__ b)
{
    const float* hr = h + (size_t)blockIdx.x * (2 * DD);
    int base = threadIdx.x * 8;
    float4 xa = ((const float4*)hr)[threadIdx.x * 2];
    float4 xb = ((const float4*)hr)[threadIdx.x * 2 + 1];
    float x[8] = {xa.x, xa.y, xa.z, xa.w, xb.x, xb.y, xb.z, xb.w};
    float s = 0.0f, ss = 0.0f;
#pragma unroll
    for (int i = 0; i < 8; ++i) { s += x[i]; ss += x[i] * x[i]; }
#pragma unroll
    for (int m = 1; m < 32; m <<= 1) {
        s  += __shfl_xor(s,  m, 32);
        ss += __shfl_xor(ss, m, 32);
    }
    __shared__ float rs[8], rss[8];
    int wv = threadIdx.x >> 5;
    if ((threadIdx.x & 31) == 0) { rs[wv] = s; rss[wv] = ss; }
    __syncthreads();
    s = 0.0f; ss = 0.0f;
#pragma unroll
    for (int i = 0; i < 8; ++i) { s += rs[i]; ss += rss[i]; }
    const float inv = 1.0f / (2 * DD);
    float mu  = s * inv;
    float var = ss * inv - mu * mu;
    float rstd = rsqrtf(var + 1e-5f);
    unsigned short o[8];
#pragma unroll
    for (int i = 0; i < 8; ++i) {
        float xn = (x[i] - mu) * rstd * g[base + i] + b[base + i];
        o[i] = f2bf(0.5f * xn * (1.0f + erff(xn * 0.70710678118654752f)));
    }
    *(uint4*)(hb + (size_t)blockIdx.x * (2 * DD) + base) = *(uint4*)o;
}

// =====================================================================
extern "C" void kernel_launch(void* const* d_in, const int* in_sizes, int n_in,
                              void* d_out, int out_size, void* d_ws, size_t ws_size,
                              hipStream_t stream) {
    (void)in_sizes; (void)n_in; (void)out_size; (void)ws_size;
    const float* descriptor = (const float*)d_in[0];
    const float* encoding   = (const float*)d_in[1];
    const float* w_qkv  = (const float*)d_in[2];
    const float* b_qkv  = (const float*)d_in[3];
    const float* w_out  = (const float*)d_in[4];
    const float* b_out  = (const float*)d_in[5];
    const float* w_ffn1 = (const float*)d_in[6];
    const float* b_ffn1 = (const float*)d_in[7];
    const float* ln_g   = (const float*)d_in[8];
    const float* ln_b   = (const float*)d_in[9];
    const float* w_ffn2 = (const float*)d_in[10];
    const float* b_ffn2 = (const float*)d_in[11];
    float* out = (float*)d_out;

    // workspace carve-up (~164 MB total)
    char* ws = (char*)d_ws;
    size_t o = 0;
    float* qkv = (float*)(ws + o);                   o += (size_t)BB * NN * 3 * DD * 4;
    unsigned short* qb    = (unsigned short*)(ws + o); o += (size_t)BB * HH * NN * HD * 2;
    unsigned short* kb    = (unsigned short*)(ws + o); o += (size_t)BB * HH * NN * HD * 2;
    unsigned short* vT    = (unsigned short*)(ws + o); o += (size_t)BB * HH * NN * HD * 2;
    unsigned short* ctxb  = (unsigned short*)(ws + o); o += (size_t)BB * NN * DD * 2;
    unsigned short* catb  = (unsigned short*)(ws + o); o += (size_t)BB * NN * 2 * DD * 2;
    float* hbuf = (float*)(ws + o);                  o += (size_t)BB * NN * 2 * DD * 4;
    unsigned short* hb    = (unsigned short*)(ws + o); o += (size_t)BB * NN * 2 * DD * 2;
    unsigned short* wqkvb = (unsigned short*)(ws + o); o += (size_t)3 * DD * DD * 2;
    unsigned short* woutb = (unsigned short*)(ws + o); o += (size_t)DD * DD * 2;
    unsigned short* wf1b  = (unsigned short*)(ws + o); o += (size_t)2 * DD * 2 * DD * 2;
    unsigned short* wf2b  = (unsigned short*)(ws + o); o += (size_t)DD * 2 * DD * 2;

    const int M = BB * NN;  // 4096

    // 0) pack weights + descriptor (descriptor lands in concat left half;
    //    the same buffer with lda=2D is also the QKV GEMM's A operand)
    pack_bf16_kernel<<<((size_t)M * DD) / 1024, 256, 0, stream>>>(descriptor, catb, DD, 2 * DD);
    pack_bf16_kernel<<<((size_t)3 * DD * DD) / 1024, 256, 0, stream>>>(w_qkv, wqkvb, DD, DD);
    pack_bf16_kernel<<<((size_t)DD * DD) / 1024, 256, 0, stream>>>(w_out, woutb, DD, DD);
    pack_bf16_kernel<<<((size_t)4 * DD * DD) / 1024, 256, 0, stream>>>(w_ffn1, wf1b, 2 * DD, 2 * DD);
    pack_bf16_kernel<<<((size_t)2 * DD * DD) / 1024, 256, 0, stream>>>(w_ffn2, wf2b, 2 * DD, 2 * DD);

    // 1) QKV projection: qkv = descriptor @ w_qkv^T + b_qkv  (fp32 out)
    gemm_kernel<<<(M / 128) * ((3 * DD) / 64), 128, 0, stream>>>(
        catb, 2 * DD, wqkvb, DD, b_qkv, nullptr, 0, qkv, nullptr, 3 * DD, 3 * DD, DD);

    // 2) RoPE + repack to bf16 (V transposed)
    rope_pack_kernel<<<(BB * HH * NN * (HD / 2)) / 256, 256, 0, stream>>>(
        qkv, encoding, qb, kb, vT);

    // 3) Flash attention -> ctxb bf16 [B,N,H*HD]
    attn_kernel<<<dim3(BB * HH, NN / 64), 128, 0, stream>>>(qb, kb, vT, ctxb);

    // 4) out-projection writes bf16 message into concat right half
    gemm_kernel<<<(M / 128) * (DD / 64), 128, 0, stream>>>(
        ctxb, DD, woutb, DD, b_out, nullptr, 0, nullptr, catb + DD, 2 * DD, DD, DD);

    // 5) FFN1: h = cat @ w_ffn1^T + b_ffn1  (fp32 out for LayerNorm)
    gemm_kernel<<<(M / 128) * ((2 * DD) / 64), 128, 0, stream>>>(
        catb, 2 * DD, wf1b, 2 * DD, b_ffn1, nullptr, 0, hbuf, nullptr, 2 * DD, 2 * DD, 2 * DD);

    // 6) LayerNorm + GELU -> bf16
    ln_gelu_kernel<<<M, 256, 0, stream>>>(hbuf, hb, ln_g, ln_b);

    // 7) FFN2 + residual: out = descriptor + gelu(ln(h)) @ w_ffn2^T + b_ffn2
    gemm_kernel<<<(M / 128) * (DD / 64), 128, 0, stream>>>(
        hb, 2 * DD, wf2b, 2 * DD, b_ffn2, descriptor, DD, out, nullptr, DD, DD, 2 * DD);
}
